// EncoderLayer_20469814132930
// MI455X (gfx1250) — matrix operand order repaired and checked by reading, not compiled, not run on hardware
//
#include <hip/hip_runtime.h>

// Problem constants (from reference)
#define DIMD   1024
#define HEADS  16
#define HDIM   64
#define FFND   4096
#define BATCH  8
#define SEQ    1024
#define MROWS  (BATCH*SEQ)          // 8192

typedef _Float16 v16h __attribute__((ext_vector_type(16)));
typedef _Float16 h8   __attribute__((ext_vector_type(8)));
typedef _Float16 h4   __attribute__((ext_vector_type(4)));
typedef float    v8f  __attribute__((ext_vector_type(8)));

union H16 { v16h v; h8 h[2]; };

// ---------------------------------------------------------------------------
// Tiled WMMA GEMM: C = op(A[M,K] x B) (+bias) (+res) (relu?)
//   BMODE==0 : B is [K,N] row-major (weights)        -> transposed into LDS
//   BMODE==1 : B is [N,K] row-major (compute A*B^T)  -> copied directly
// Block tile: 128 x BN, K-step 32. 256 threads = 8 waves (4 x 2), each wave
// owns a 32 x (NT*16) accumulator patch of 16x16 WMMA tiles.
// fp32 global -> f16 LDS conversion on the fly; fp32 accumulation via
// v_wmma_f32_16x16x32_f16. Register double-buffering overlaps the next
// K-tile's global loads with the current tile's WMMA work. All staging loops
// have uniform, compile-time trip counts (no exec masking anywhere).
// Batched via blockIdx.z: offset = (z/Hin)*SB + (z%Hin)*SH per operand.
// ---------------------------------------------------------------------------
template<int BN, int NT, int BMODE>
__global__ __launch_bounds__(256)
void gemm_wmma(const float* __restrict__ A,  int lda, long aSB, long aSH,
               const float* __restrict__ Bm, int ldb, long bSB, long bSH,
               float* __restrict__ C,        int ldc, long cSB, long cSH,
               int Kdim, int Hin,
               const float* __restrict__ bias,
               const float* __restrict__ res, int ldres,
               int relu)
{
    constexpr int BM  = 128;
    constexpr int BK  = 32;
    constexpr int BKP = 40;   // padded row stride (halves): 80B -> conflict-free frags
    constexpr int AIT = (BM * BK / 4) / 256;   // 4 float4 loads per thread
    constexpr int BIT = (BN * BK / 4) / 256;   // 4 (BN=128) or 2 (BN=64)
    constexpr int NB4 = BN / 4;

    __shared__ __align__(16) _Float16 sA[BM * BKP];
    __shared__ __align__(16) _Float16 sB[BN * BKP];

    const int z  = blockIdx.z;
    const int zo = z / Hin;
    const int zi = z - zo * Hin;
    A  += (long)zo * aSB + (long)zi * aSH;
    Bm += (long)zo * bSB + (long)zi * bSH;
    C  += (long)zo * cSB + (long)zi * cSH;

    const int m0 = blockIdx.y * BM;
    const int n0 = blockIdx.x * BN;

    const int tid   = threadIdx.x;
    const int lane  = tid & 31;
    const int wave  = tid >> 5;
    const int waveM = wave & 3;        // 4 wave-rows
    const int waveN = wave >> 2;       // 2 wave-cols
    const int l16   = lane & 15;
    const int hi    = lane >> 4;       // half-wave select

    v8f acc[2][NT];
    const v8f vzero = {0.f,0.f,0.f,0.f,0.f,0.f,0.f,0.f};
    #pragma unroll
    for (int i = 0; i < 2; ++i)
        #pragma unroll
        for (int j = 0; j < NT; ++j) acc[i][j] = vzero;

    // register tile buffers (double-buffer between K-steps)
    float4 ra[AIT], rb[BIT];

    auto loadA = [&](int k0) {
        #pragma unroll
        for (int j = 0; j < AIT; ++j) {
            const int i  = tid + j * 256;
            const int r  = i >> 3;          // BK/4 == 8
            const int c4 = i & 7;
            ra[j] = *(const float4*)(A + (long)(m0 + r) * lda + k0 + c4 * 4);
        }
    };
    auto loadB = [&](int k0) {
        if (BMODE == 1) {
            #pragma unroll
            for (int j = 0; j < BIT; ++j) {
                const int i  = tid + j * 256;
                const int r  = i >> 3;
                const int c4 = i & 7;
                rb[j] = *(const float4*)(Bm + (long)(n0 + r) * ldb + k0 + c4 * 4);
            }
        } else {
            #pragma unroll
            for (int j = 0; j < BIT; ++j) {
                const int i  = tid + j * 256;
                const int r  = i / NB4;     // k within tile
                const int c4 = i - r * NB4; // n/4 within tile
                rb[j] = *(const float4*)(Bm + (long)(k0 + r) * ldb + n0 + c4 * 4);
            }
        }
    };
    auto storeA = [&]() {
        #pragma unroll
        for (int j = 0; j < AIT; ++j) {
            const int i  = tid + j * 256;
            const int r  = i >> 3;
            const int c4 = i & 7;
            h4 hh; hh.x = (_Float16)ra[j].x; hh.y = (_Float16)ra[j].y;
                   hh.z = (_Float16)ra[j].z; hh.w = (_Float16)ra[j].w;
            *(h4*)(sA + r * BKP + c4 * 4) = hh;
        }
    };
    auto storeB = [&]() {
        if (BMODE == 1) {
            #pragma unroll
            for (int j = 0; j < BIT; ++j) {
                const int i  = tid + j * 256;
                const int r  = i >> 3;
                const int c4 = i & 7;
                h4 hh; hh.x = (_Float16)rb[j].x; hh.y = (_Float16)rb[j].y;
                       hh.z = (_Float16)rb[j].z; hh.w = (_Float16)rb[j].w;
                *(h4*)(sB + r * BKP + c4 * 4) = hh;
            }
        } else {
            #pragma unroll
            for (int j = 0; j < BIT; ++j) {
                const int i  = tid + j * 256;
                const int r  = i / NB4;
                const int c4 = i - r * NB4;
                sB[(c4 * 4 + 0) * BKP + r] = (_Float16)rb[j].x;
                sB[(c4 * 4 + 1) * BKP + r] = (_Float16)rb[j].y;
                sB[(c4 * 4 + 2) * BKP + r] = (_Float16)rb[j].z;
                sB[(c4 * 4 + 3) * BKP + r] = (_Float16)rb[j].w;
            }
        }
    };

    loadA(0);
    loadB(0);

    for (int k0 = 0; k0 < Kdim; k0 += BK) {
        storeA();
        storeB();
        __syncthreads();

        // prefetch next K-tile into registers; loads overlap the WMMAs below
        if (k0 + BK < Kdim) { loadA(k0 + BK); loadB(k0 + BK); }

        // ---- A fragments (16x32 f16 layout, ISA 7.12.2):
        // lanes 0-15: row=l16, K 0..7 then 16..23 ; lanes 16-31: K 8..15, 24..31
        v16h aF[2];
        #pragma unroll
        for (int tm = 0; tm < 2; ++tm) {
            const int row = waveM * 32 + tm * 16 + l16;
            H16 u;
            u.h[0] = *(const h8*)(sA + row * BKP + hi * 8);
            u.h[1] = *(const h8*)(sA + row * BKP + 16 + hi * 8);
            aF[tm] = u.v;
        }
        // ---- B fragments + WMMA ----
        #pragma unroll
        for (int tn = 0; tn < NT; ++tn) {
            const int col = waveN * (NT * 16) + tn * 16 + l16;
            H16 u;
            u.h[0] = *(const h8*)(sB + col * BKP + hi * 8);
            u.h[1] = *(const h8*)(sB + col * BKP + 16 + hi * 8);
            const v16h bF = u.v;
            #pragma unroll
            for (int tm = 0; tm < 2; ++tm) {
                acc[tm][tn] = __builtin_amdgcn_wmma_f32_16x16x32_f16(
                    false, aF[tm], false, bF, (short)0, acc[tm][tn], false, false);
            }
        }
        __syncthreads();
    }

    // ---- epilogue: D layout (7.12.2): VGPR r -> M = hi*8 + r, N = l16 ----
    #pragma unroll
    for (int tn = 0; tn < NT; ++tn) {
        const int col = n0 + waveN * (NT * 16) + tn * 16 + l16;
        const float bv = bias ? bias[col] : 0.0f;
        #pragma unroll
        for (int tm = 0; tm < 2; ++tm) {
            #pragma unroll
            for (int r = 0; r < 8; ++r) {
                const int row = m0 + waveM * 32 + tm * 16 + hi * 8 + r;
                float v = acc[tm][tn][r] + bv;
                if (res)  v += res[(long)row * ldres + col];
                if (relu) v = fmaxf(v, 0.0f);
                C[(long)row * ldc + col] = v;
            }
        }
    }
}

// ---------------------------------------------------------------------------
// Row softmax over SEQ keys, with 1/sqrt(HD) scale and additive mask.
// One 256-thread block per row; in-place on the attn-weights output region.
// ---------------------------------------------------------------------------
__global__ __launch_bounds__(256)
void softmax_kernel(float* __restrict__ attn, const float* __restrict__ mask)
{
    const long row = blockIdx.x;                 // [0, B*H*S)
    const int  b   = (int)(row >> 14);           // / (H*S) = 16384
    float* p = attn + row * SEQ;

    const int tid  = threadIdx.x;
    const int lane = tid & 31;
    const int wave = tid >> 5;
    __shared__ float red[8];

    float4 f = *(const float4*)(p + tid * 4);
    const float* mrow = mask + (long)b * SEQ + tid * 4;
    const float sc = 0.125f;                     // 1/sqrt(64)
    float v0 = f.x * sc + (1.0f - mrow[0]) * -10000.0f;
    float v1 = f.y * sc + (1.0f - mrow[1]) * -10000.0f;
    float v2 = f.z * sc + (1.0f - mrow[2]) * -10000.0f;
    float v3 = f.w * sc + (1.0f - mrow[3]) * -10000.0f;

    float mx = fmaxf(fmaxf(v0, v1), fmaxf(v2, v3));
    #pragma unroll
    for (int o = 16; o > 0; o >>= 1) mx = fmaxf(mx, __shfl_down(mx, o, 32));
    if (lane == 0) red[wave] = mx;
    __syncthreads();
    if (tid == 0) { float t = red[0]; for (int i = 1; i < 8; ++i) t = fmaxf(t, red[i]); red[0] = t; }
    __syncthreads();
    mx = red[0];
    __syncthreads();

    float e0 = __expf(v0 - mx), e1 = __expf(v1 - mx);
    float e2 = __expf(v2 - mx), e3 = __expf(v3 - mx);
    float s = e0 + e1 + e2 + e3;
    #pragma unroll
    for (int o = 16; o > 0; o >>= 1) s += __shfl_down(s, o, 32);
    if (lane == 0) red[wave] = s;
    __syncthreads();
    if (tid == 0) { float t = 0.f; for (int i = 0; i < 8; ++i) t += red[i]; red[0] = t; }
    __syncthreads();
    const float inv = 1.0f / red[0];

    f.x = e0 * inv; f.y = e1 * inv; f.z = e2 * inv; f.w = e3 * inv;
    *(float4*)(p + tid * 4) = f;
}

// ---------------------------------------------------------------------------
// LayerNorm over D=1024: out = (x(+res) - mu) * rsqrt(var+eps) * g + b.
// One 256-thread block per row; in-place safe (reads precede barrier).
// ---------------------------------------------------------------------------
__global__ __launch_bounds__(256)
void layernorm_kernel(const float* __restrict__ x, const float* __restrict__ res,
                      const float* __restrict__ g, const float* __restrict__ be,
                      float eps, float* __restrict__ out)
{
    const long row = blockIdx.x;
    const int tid  = threadIdx.x;
    const int lane = tid & 31;
    const int wave = tid >> 5;
    __shared__ float redS[8], redQ[8];

    float4 v = *(const float4*)(x + row * DIMD + tid * 4);
    if (res) {
        float4 r = *(const float4*)(res + row * DIMD + tid * 4);
        v.x += r.x; v.y += r.y; v.z += r.z; v.w += r.w;
    }
    float s = v.x + v.y + v.z + v.w;
    float q = v.x * v.x + v.y * v.y + v.z * v.z + v.w * v.w;
    #pragma unroll
    for (int o = 16; o > 0; o >>= 1) { s += __shfl_down(s, o, 32); q += __shfl_down(q, o, 32); }
    if (lane == 0) { redS[wave] = s; redQ[wave] = q; }
    __syncthreads();
    if (tid == 0) {
        float ts = 0.f, tq = 0.f;
        for (int i = 0; i < 8; ++i) { ts += redS[i]; tq += redQ[i]; }
        redS[0] = ts; redQ[0] = tq;
    }
    __syncthreads();
    const float mu  = redS[0] * (1.0f / DIMD);
    const float var = redQ[0] * (1.0f / DIMD) - mu * mu;
    const float rs  = rsqrtf(var + eps);

    const int c = tid * 4;
    float4 o;
    o.x = (v.x - mu) * rs * g[c + 0] + be[c + 0];
    o.y = (v.y - mu) * rs * g[c + 1] + be[c + 1];
    o.z = (v.z - mu) * rs * g[c + 2] + be[c + 2];
    o.w = (v.w - mu) * rs * g[c + 3] + be[c + 3];
    *(float4*)(out + row * DIMD + tid * 4) = o;
}

// ---------------------------------------------------------------------------
extern "C" void kernel_launch(void* const* d_in, const int* in_sizes, int n_in,
                              void* d_out, int out_size, void* d_ws, size_t ws_size,
                              hipStream_t stream)
{
    (void)in_sizes; (void)n_in; (void)out_size; (void)ws_size;

    const float* query = (const float*)d_in[0];
    const float* key   = (const float*)d_in[1];
    const float* value = (const float*)d_in[2];
    const float* amask = (const float*)d_in[3];
    const float* wq = (const float*)d_in[4];  const float* bq = (const float*)d_in[5];
    const float* wk = (const float*)d_in[6];  const float* bk = (const float*)d_in[7];
    const float* wv = (const float*)d_in[8];  const float* bv = (const float*)d_in[9];
    const float* wo = (const float*)d_in[10]; const float* bo = (const float*)d_in[11];
    const float* ln_mha_g  = (const float*)d_in[12]; const float* ln_mha_b  = (const float*)d_in[13];
    const float* w1 = (const float*)d_in[14]; const float* b1 = (const float*)d_in[15];
    const float* w2 = (const float*)d_in[16]; const float* b2 = (const float*)d_in[17];
    const float* ln_attn_g = (const float*)d_in[18]; const float* ln_attn_b = (const float*)d_in[19];
    const float* ln_ffn_g  = (const float*)d_in[20]; const float* ln_ffn_b  = (const float*)d_in[21];

    const long MD = (long)MROWS * DIMD;          // 8,388,608
    float* ws  = (float*)d_ws;
    float* Q   = ws;
    float* Kb  = ws + 1 * MD;
    float* Vb  = ws + 2 * MD;
    float* CTX = ws + 3 * MD;
    float* T1  = ws + 4 * MD;
    float* A2  = ws + 5 * MD;
    float* FFN = ws + 6 * MD;
    float* H1  = ws + 7 * MD;                    // 8192 x 4096 = 4*MD

    float* outp = (float*)d_out;                 // [B,S,D]
    float* attn = outp + MD;                     // [B,H,S,S]

    const dim3 blk(256);
    const long SD = (long)SEQ * DIMD;
    const long SS = (long)SEQ * SEQ;

    // 1-3: Q/K/V projections: [8192,1024] @ [1024,1024] + bias
    gemm_wmma<128,4,0><<<dim3(DIMD/128, MROWS/128, 1), blk, 0, stream>>>(
        query, DIMD, 0, 0, wq, DIMD, 0, 0, Q, DIMD, 0, 0, DIMD, 1, bq, nullptr, 0, 0);
    gemm_wmma<128,4,0><<<dim3(DIMD/128, MROWS/128, 1), blk, 0, stream>>>(
        key,   DIMD, 0, 0, wk, DIMD, 0, 0, Kb, DIMD, 0, 0, DIMD, 1, bk, nullptr, 0, 0);
    gemm_wmma<128,4,0><<<dim3(DIMD/128, MROWS/128, 1), blk, 0, stream>>>(
        value, DIMD, 0, 0, wv, DIMD, 0, 0, Vb, DIMD, 0, 0, DIMD, 1, bv, nullptr, 0, 0);

    // 4: scores = Q_bh @ K_bh^T per (b,h): M=N=1024, K=64  (B given as [N,K])
    gemm_wmma<128,4,1><<<dim3(SEQ/128, SEQ/128, BATCH*HEADS), blk, 0, stream>>>(
        Q,  DIMD, SD, HDIM,
        Kb, DIMD, SD, HDIM,
        attn, SEQ, (long)HEADS * SS, SS,
        HDIM, HEADS, nullptr, nullptr, 0, 0);

    // 5: softmax (scale + mask) in-place -> final attention weights output
    softmax_kernel<<<dim3(BATCH*HEADS*SEQ), blk, 0, stream>>>(attn, amask);

    // 6: context = attn @ V_bh : M=1024, K=1024, N=64 per (b,h)
    gemm_wmma<64,2,0><<<dim3(HDIM/64, SEQ/128, BATCH*HEADS), blk, 0, stream>>>(
        attn, SEQ, (long)HEADS * SS, SS,
        Vb,  DIMD, SD, HDIM,
        CTX, DIMD, SD, HDIM,
        SEQ, HEADS, nullptr, nullptr, 0, 0);

    // 7: T1 = context @ wo + bo + query
    gemm_wmma<128,4,0><<<dim3(DIMD/128, MROWS/128, 1), blk, 0, stream>>>(
        CTX, DIMD, 0, 0, wo, DIMD, 0, 0, T1, DIMD, 0, 0, DIMD, 1, bo, query, DIMD, 0);

    // 8: T1 = LN(T1), eps=1e-8 (MHA-internal LN), in-place
    layernorm_kernel<<<dim3(MROWS), blk, 0, stream>>>(
        T1, nullptr, ln_mha_g, ln_mha_b, 1e-8f, T1);

    // 9: A2 = LN(query + T1), eps=1e-6
    layernorm_kernel<<<dim3(MROWS), blk, 0, stream>>>(
        T1, query, ln_attn_g, ln_attn_b, 1e-6f, A2);

    // 10: H1 = relu(A2 @ w1 + b1) : [8192,1024] @ [1024,4096]
    gemm_wmma<128,4,0><<<dim3(FFND/128, MROWS/128, 1), blk, 0, stream>>>(
        A2, DIMD, 0, 0, w1, FFND, 0, 0, H1, FFND, 0, 0, DIMD, 1, b1, nullptr, 0, 1);

    // 11: FFN = H1 @ w2 + b2 + A2 : [8192,4096] @ [4096,1024]
    gemm_wmma<128,4,0><<<dim3(DIMD/128, MROWS/128, 1), blk, 0, stream>>>(
        H1, FFND, 0, 0, w2, DIMD, 0, 0, FFN, DIMD, 0, 0, FFND, 1, b2, A2, DIMD, 0);

    // 12: out = LN(FFN), eps=1e-6
    layernorm_kernel<<<dim3(MROWS), blk, 0, stream>>>(
        FFN, nullptr, ln_ffn_g, ln_ffn_b, 1e-6f, outp);
}
